// ProbSparseSelfAttention_51874615001712
// MI455X (gfx1250) — matrix operand order repaired and checked by reading, not compiled, not run on hardware
//
#include <hip/hip_runtime.h>

// ---------------------------------------------------------------------------
// ProbSparse self-attention, MI455X / gfx1250 (wave32, WMMA bf16)
// B=2, L=2048, DIM=512, H=8, Dh=64, top_k=512
//
// Workspace layout (bf16 elements):
//   x_bf    [2*2048*512]      staged x
//   wqkv_bf [1536*512]        staged w_qkv
//   wout_bf [512*512]         staged w_out
//   q_bf    [16][2048][64]    per (b,h)
//   k_bf    [16][2048][64]
//   vT_bf   [16][64][2048]    V transposed for contiguous B-fragments
//   attn_bf [2*2048*512]      attention output (head-concat)
// total ~23 MB
// ---------------------------------------------------------------------------

typedef __attribute__((ext_vector_type(8)))  __bf16 v8bf;
typedef __attribute__((ext_vector_type(16))) __bf16 v16bf;
typedef __attribute__((ext_vector_type(8)))  float  v8f;

#define Bc   2
#define Lc   2048
#define Dc   512
#define Hc   8
#define Dh   64
#define TOPK 512

static __device__ inline v16bf cat8(v8bf lo, v8bf hi) {
  v16bf r;
#pragma unroll
  for (int i = 0; i < 8; ++i) { r[i] = lo[i]; r[8 + i] = hi[i]; }
  return r;
}

// A fragment (16x32, MxK): lane l holds row M=l&15, K = base..base+7 and
// base+16..base+23, base = (l>>4)*8.  rowp points at A[row][kk..].
static __device__ inline v16bf load_afrag(const __bf16* rowp, int lane, int kk) {
  const int base = kk + ((lane >> 4) << 3);
  v8bf lo = *(const v8bf*)(rowp + base);
  v8bf hi = *(const v8bf*)(rowp + base + 16);
  return cat8(lo, hi);
}

// B fragment (32x16, KxN): lane l holds column N=l&15, K = base..base+7 and
// base+16..base+23, base = (l>>4)*8.  colrow points at the memory row serving as column N
// (all our GEMMs are X*W^T shaped, so B's column n is a contiguous row).
static __device__ inline v16bf load_bfrag(const __bf16* colrow, int lane, int kk) {
  const int base = kk + ((lane >> 4) << 3);
  v8bf lo = *(const v8bf*)(colrow + base);
  v8bf hi = *(const v8bf*)(colrow + base + 16);
  return cat8(lo, hi);
}

static __device__ inline v8f wmma_bf16(v16bf a, v16bf b, v8f c) {
  return __builtin_amdgcn_wmma_f32_16x16x32_bf16(
      /*neg_a=*/false, a, /*neg_b=*/false, b,
      /*c_mod=*/(short)0, c, /*reuse_a=*/false, /*reuse_b=*/false);
}

// order-preserving float <-> uint key (total order, NaN-free inputs)
static __device__ inline unsigned f2k(float f) {
  unsigned b = __float_as_uint(f);
  return (b & 0x80000000u) ? ~b : (b | 0x80000000u);
}
static __device__ inline float k2f(unsigned k) {
  unsigned b = (k & 0x80000000u) ? (k ^ 0x80000000u) : ~k;
  return __uint_as_float(b);
}

// ---------------------------------------------------------------------------
__global__ void cvt_f32_bf16(const float* __restrict__ s, __bf16* __restrict__ d, int n) {
  int i = blockIdx.x * blockDim.x + threadIdx.x;
  int stride = gridDim.x * blockDim.x;
  for (; i < n; i += stride) d[i] = (__bf16)s[i];
}

// ---------------------------------------------------------------------------
// QKV projection: qkv[m][n] = sum_k x[m][k]*w_qkv[n][k] + b_qkv[n]
// M = B*L = 4096 (256 tiles), N = 1536 (24 groups of 64).
// One wave computes a 16x64 tile: 4 accumulators, A-fragment reused 4x.
__global__ __launch_bounds__(256) void qkv_gemm(
    const __bf16* __restrict__ xb, const __bf16* __restrict__ wb,
    const float* __restrict__ bias,
    __bf16* __restrict__ qb, __bf16* __restrict__ kb, __bf16* __restrict__ vtb) {
  const int lane = threadIdx.x & 31, wave = threadIdx.x >> 5;
  const int job = blockIdx.x * 8 + wave;        // 6144 jobs
  const int mt = job / 24, ng = job % 24;
  const __bf16* arow = xb + (mt * 16 + (lane & 15)) * Dc;
  const __bf16* brow = wb + (ng * 64 + (lane & 15)) * Dc;

  v8f c[4] = {};
#pragma unroll
  for (int kk = 0; kk < Dc; kk += 32) {
    const v16bf a = load_afrag(arow, lane, kk);
#pragma unroll
    for (int j = 0; j < 4; ++j)
      c[j] = wmma_bf16(a, load_bfrag(brow + j * 16 * Dc, lane, kk), c[j]);
  }

#pragma unroll
  for (int j = 0; j < 4; ++j) {
    const int n = ng * 64 + j * 16 + (lane & 15);
    const float bv = bias[n];
    const int t = n >> 9;          // 0=q, 1=k, 2=v
    const int h = (n >> 6) & 7;
    const int d = n & 63;
#pragma unroll
    for (int r = 0; r < 8; ++r) {
      const int m = mt * 16 + r + ((lane >> 4) << 3);
      const int bidx = m >> 11;    // batch
      const int l = m & (Lc - 1);
      const __bf16 o = (__bf16)(c[j][r] + bv);
      const int bh = bidx * Hc + h;
      if (t == 0)      qb[(bh * Lc + l) * Dh + d] = o;
      else if (t == 1) kb[(bh * Lc + l) * Dh + d] = o;
      else             vtb[(bh * Dh + d) * Lc + l] = o;   // transposed
    }
  }
}

// ---------------------------------------------------------------------------
// Attention core: one block = one (b,h) x one 16-query tile.
// LDS: sc[16][2048] f32 (128KB) + pb[16][2048] bf16 (64KB) = 192KB dynamic.
__global__ __launch_bounds__(256) void attn_kernel(
    const __bf16* __restrict__ qb, const __bf16* __restrict__ kb,
    const __bf16* __restrict__ vtb, __bf16* __restrict__ attn) {
  extern __shared__ char smem[];
  float*  sc = (float*)smem;                        // 16 x 2048 scores
  __bf16* pb = (__bf16*)(smem + 16 * Lc * 4);       // 16 x 2048 probs

  const int lane = threadIdx.x & 31, wave = threadIdx.x >> 5;
  const int bh = blockIdx.x >> 7;                   // (b*8+h), 16 of them
  const int qt = blockIdx.x & 127;
  const int l0 = qt << 4;

  // ---- phase 1: S = scale * Q_tile @ K^T  (16 x 2048 strip into LDS) ----
  const __bf16* qrow = qb + (bh * Lc + l0 + (lane & 15)) * Dh;
  const v16bf a0 = load_afrag(qrow, lane, 0);
  const v16bf a1 = load_afrag(qrow, lane, 32);
#pragma unroll 1
  for (int j = 0; j < 16; ++j) {
    const int n0 = (wave * 16 + j) << 4;
    const __bf16* krow = kb + (bh * Lc + n0 + (lane & 15)) * Dh;
    v8f c = {};
    c = wmma_bf16(a0, load_bfrag(krow, lane, 0), c);
    c = wmma_bf16(a1, load_bfrag(krow, lane, 32), c);
#pragma unroll
    for (int r = 0; r < 8; ++r) {
      const int m = r + ((lane >> 4) << 3);
      sc[m * Lc + n0 + (lane & 15)] = c[r] * 0.125f;  // 1/sqrt(64)
    }
  }
  __syncthreads();

  // ---- phase 2+3: exact top-k threshold + masked softmax, 2 rows/wave ----
#pragma unroll 1
  for (int rr = 0; rr < 2; ++rr) {
    const int row = wave + rr * 8;
    unsigned keys[64];
#pragma unroll
    for (int i = 0; i < 64; ++i) keys[i] = f2k(sc[row * Lc + i * 32 + lane]);

    unsigned km = 0;
#pragma unroll
    for (int i = 0; i < 64; ++i) km = keys[i] > km ? keys[i] : km;
    for (int off = 16; off; off >>= 1) {
      unsigned o = (unsigned)__shfl_xor((int)km, off, 32);
      km = o > km ? o : km;
    }

    // bitwise binary search: largest T with count(key >= T) >= TOPK
    unsigned T = 0;
    for (int bit = 31; bit >= 0; --bit) {
      const unsigned cand = T | (1u << bit);
      int cnt = 0;
#pragma unroll
      for (int i = 0; i < 64; ++i) cnt += (keys[i] >= cand) ? 1 : 0;
      for (int off = 16; off; off >>= 1) cnt += __shfl_xor(cnt, off, 32);
      if (cnt >= TOPK) T = cand;
    }

    const float fmax = k2f(km);
    float s = 0.f;
#pragma unroll
    for (int i = 0; i < 64; ++i)
      if (keys[i] >= T) s += __expf(k2f(keys[i]) - fmax);
    for (int off = 16; off; off >>= 1) s += __shfl_xor(s, off, 32);
    const float inv = 1.f / s;
#pragma unroll
    for (int i = 0; i < 64; ++i) {
      const float p = (keys[i] >= T) ? __expf(k2f(keys[i]) - fmax) * inv : 0.f;
      pb[row * Lc + i * 32 + lane] = (__bf16)p;
    }
  }
  __syncthreads();

  // ---- phase 4: O = P @ V  (16 x 64).
  // Each wave takes a K-eighth (256) across all 4 head-dim column tiles:
  // one LDS A-fragment feeds 4 wmmas. Partials reduced through dead sc strip.
  const __bf16* prow = pb + (lane & 15) * Lc + wave * 256;
  const __bf16* vbase = vtb + (bh * Dh + (lane & 15)) * Lc + wave * 256;
  v8f c[4] = {};
#pragma unroll
  for (int kk = 0; kk < 256; kk += 32) {
    const v16bf a = load_afrag(prow, lane, kk);
#pragma unroll
    for (int j = 0; j < 4; ++j)
      c[j] = wmma_bf16(a, load_bfrag(vbase + j * 16 * Lc, lane, kk), c[j]);
  }

  float* part = sc;  // scores strip is dead; 8 waves x 4 tiles x 256 f32 = 32KB
#pragma unroll
  for (int j = 0; j < 4; ++j)
#pragma unroll
    for (int r = 0; r < 8; ++r)
      part[(wave * 4 + j) * 256 + r * 32 + lane] = c[j][r];
  __syncthreads();

  if (wave < 4) {
    const int j = wave;            // this wave finalizes column tile j
    const int b_ = bh >> 3, h = bh & 7;
#pragma unroll
    for (int r = 0; r < 8; ++r) {
      float v = 0.f;
#pragma unroll
      for (int w8 = 0; w8 < 8; ++w8) v += part[(w8 * 4 + j) * 256 + r * 32 + lane];
      const int m = r + ((lane >> 4) << 3);
      const int n = j * 16 + (lane & 15);
      attn[(b_ * Lc + l0 + m) * Dc + h * Dh + n] = (__bf16)v;
    }
  }
}

// ---------------------------------------------------------------------------
// Output projection: out[m][n] = sum_k attn[m][k]*w_out[n][k] + b_out[n]
// M = 4096 (256 tiles), N = 512 (8 groups of 64); 16x64 tile per wave.
__global__ __launch_bounds__(256) void out_gemm(
    const __bf16* __restrict__ ab, const __bf16* __restrict__ wob,
    const float* __restrict__ bias, float* __restrict__ out) {
  const int lane = threadIdx.x & 31, wave = threadIdx.x >> 5;
  const int job = blockIdx.x * 8 + wave;        // 2048 jobs
  const int mt = job / 8, ng = job % 8;
  const __bf16* arow = ab + (mt * 16 + (lane & 15)) * Dc;
  const __bf16* brow = wob + (ng * 64 + (lane & 15)) * Dc;

  v8f c[4] = {};
#pragma unroll
  for (int kk = 0; kk < Dc; kk += 32) {
    const v16bf a = load_afrag(arow, lane, kk);
#pragma unroll
    for (int j = 0; j < 4; ++j)
      c[j] = wmma_bf16(a, load_bfrag(brow + j * 16 * Dc, lane, kk), c[j]);
  }

#pragma unroll
  for (int j = 0; j < 4; ++j) {
    const int n = ng * 64 + j * 16 + (lane & 15);
    const float bv = bias[n];
#pragma unroll
    for (int r = 0; r < 8; ++r) {
      const int m = mt * 16 + r + ((lane >> 4) << 3);
      out[m * Dc + n] = c[j][r] + bv;
    }
  }
}

// ---------------------------------------------------------------------------
extern "C" void kernel_launch(void* const* d_in, const int* in_sizes, int n_in,
                              void* d_out, int out_size, void* d_ws, size_t ws_size,
                              hipStream_t stream) {
  const float* x     = (const float*)d_in[0];   // [2,2048,512]
  const float* w_qkv = (const float*)d_in[1];   // [1536,512]
  const float* b_qkv = (const float*)d_in[2];   // [1536]
  const float* w_out = (const float*)d_in[3];   // [512,512]
  const float* b_out = (const float*)d_in[4];   // [512]
  float* out = (float*)d_out;

  const int XN  = Bc * Lc * Dc;      // 2097152
  const int WQN = 3 * Dc * Dc;       // 786432
  const int WON = Dc * Dc;           // 262144
  const int QN  = Bc * Hc * Lc * Dh; // 2097152

  __bf16* xb   = (__bf16*)d_ws;
  __bf16* wqb  = xb + XN;
  __bf16* wob  = wqb + WQN;
  __bf16* qb   = wob + WON;
  __bf16* kb   = qb + QN;
  __bf16* vtb  = kb + QN;
  __bf16* ab   = vtb + QN;           // total ~23 MB of d_ws

  cvt_f32_bf16<<<1024, 256, 0, stream>>>(x, xb, XN);
  cvt_f32_bf16<<<512, 256, 0, stream>>>(w_qkv, wqb, WQN);
  cvt_f32_bf16<<<256, 256, 0, stream>>>(w_out, wob, WON);

  // 256 M-tiles * 24 N-groups = 6144 wave-jobs / 8 waves per block
  qkv_gemm<<<768, 256, 0, stream>>>(xb, wqb, b_qkv, qb, kb, vtb);

  // 16 (b,h) * 128 query tiles; 192 KB dynamic LDS per block
  attn_kernel<<<2048, 256, 16 * Lc * 4 + 16 * Lc * 2, stream>>>(qb, kb, vtb, ab);

  // 256 M-tiles * 8 N-groups = 2048 wave-jobs / 8 waves per block
  out_gemm<<<256, 256, 0, stream>>>(ab, wob, b_out, out);
}
